// DenseTripletLoss_69672959475799
// MI455X (gfx1250) — matrix operand order repaired and checked by reading, not compiled, not run on hardware
//
#include <hip/hip_runtime.h>
#include <hip/hip_bf16.h>

#define BB 8
#define CC 128
#define HC 64
#define WC 64
#define NN 4096      // HC*WC
#define HH 512
#define WW 512
#define GSZ 8
#define LDS_STRIDE 272  // 256B row + 16B pad (TDM pad_amount) to break bank conflicts

typedef __attribute__((ext_vector_type(16))) __bf16 v16bf;
typedef __attribute__((ext_vector_type(8)))  float  v8f;
typedef __attribute__((ext_vector_type(4)))  unsigned v4u;
typedef __attribute__((ext_vector_type(8)))  int      v8i;
typedef __attribute__((ext_vector_type(4)))  int      v4i;

union FragBf { uint4 u4[2]; v16bf v; };

__device__ __forceinline__ unsigned f2bf(float f) {
    union { float f; unsigned u; } v; v.f = f;
    unsigned r = v.u + 0x7FFFu + ((v.u >> 16) & 1u);   // RNE to bf16
    return r >> 16;
}

// Tensor Data Mover: load a 64-row x 256B 2D tile (4-byte elements) from a
// row-major tensor (row stride 256B) into LDS, hardware-padding +16B after
// every 256B row so LDS rows land on a 272B stride (bank-conflict free).
__device__ __forceinline__ void tdm_load_tile(unsigned lds_addr, const void* gaddr) {
    unsigned long long ga = (unsigned long long)(size_t)gaddr;
    v4u g0;
    g0[0] = 1u;                                    // count=1, user descriptor
    g0[1] = lds_addr;                              // LDS byte address
    g0[2] = (unsigned)ga;                          // global_addr[31:0]
    g0[3] = (unsigned)((ga >> 32) & 0x01FFFFFFull) // global_addr[56:32]
          | (2u << 30);                            // type=2 ("image")
    v8i g1;
    g1[0] = (int)((2u << 16)   // data_size = 4B elements
                | (1u << 20)   // pad_enable
                | (5u << 22)   // pad_interval: 64 dwords (256B)
                | (3u << 25)); // pad_amount:   4 dwords (16B)
    g1[1] = (int)(64u << 16);    // tensor_dim0 = 64 dwords   (bits 79:48)
    g1[2] = (int)(4096u << 16);  // tensor_dim1 = 4096 rows   (bits 111:80)
    g1[3] = (int)(64u << 16);    // tile_dim0   = 64 dwords   (bits 127:112)
    g1[4] = 64;                  // tile_dim1   = 64 rows     (bits 143:128)
    g1[5] = 64;                  // tensor_dim0_stride = 64 dwords (bits 207:160)
    g1[6] = 0;
    g1[7] = 0;
    v4i z4 = {0, 0, 0, 0};       // 2D tile: groups 2/3 unused
    v8i z8 = {0, 0, 0, 0, 0, 0, 0, 0};
    __builtin_amdgcn_tensor_load_to_lds(g0, g1, z4, z4, z8, 0);
}

// ---------------------------------------------------------------------------
// Kernel 1: per (b, n) warp cell centers via homo12, compute match mask,
// normalize desc1/desc2 (store bf16 row-major), and pos = 2-2<d1n, w_desc1>.
// ---------------------------------------------------------------------------
__global__ __launch_bounds__(256) void prep_kernel(
    const float* __restrict__ desc1, const float* __restrict__ desc2,
    const float* __restrict__ homo12,
    unsigned* __restrict__ d1n, unsigned* __restrict__ d2n,   // packed bf16x2
    float* __restrict__ wyA, float* __restrict__ wxA,
    float* __restrict__ matchA, float* __restrict__ posA)
{
    int idx = blockIdx.x * blockDim.x + threadIdx.x;
    if (idx >= BB * NN) return;
    int b = idx >> 12, n = idx & (NN - 1);

    const float* H = homo12 + b * 9;
    float x = (float)(n & (WC - 1)) * 8.f + 3.5f;
    float y = (float)(n >> 6) * 8.f + 3.5f;
    float wz  = H[6]*x + H[7]*y + H[8] + 1e-8f;
    float inv = 1.f / wz;
    float wx = (H[0]*x + H[1]*y + H[2]) * inv;
    float wy = (H[3]*x + H[4]*y + H[5]) * inv;
    wyA[idx] = wy; wxA[idx] = wx;
    matchA[idx] = (wy >= 0.f && wy <= (float)(HH - 1) &&
                   wx >= 0.f && wx <= (float)(WW - 1)) ? 1.f : 0.f;

    // bilinear corners in descriptor-cell space
    float cy = (wy - 3.5f) * 0.125f;
    float cx = (wx - 3.5f) * 0.125f;
    float y0f = floorf(cy), x0f = floorf(cx);
    float fy = cy - y0f, fx = cx - x0f;
    int y0 = (int)y0f, x0 = (int)x0f;
    float wq[4]; int iq[4];
    int   ys[2]  = { y0, y0 + 1 },   xs[2]  = { x0, x0 + 1 };
    float wyv[2] = { 1.f - fy, fy }, wxv[2] = { 1.f - fx, fx };
    #pragma unroll
    for (int a = 0; a < 2; a++)
    #pragma unroll
    for (int q = 0; q < 2; q++) {
        int yi = ys[a], xi = xs[q];
        bool ok = (yi >= 0 && yi < HC && xi >= 0 && xi < WC);
        int yc = yi < 0 ? 0 : (yi > HC - 1 ? HC - 1 : yi);
        int xc = xi < 0 ? 0 : (xi > WC - 1 ? WC - 1 : xi);
        iq[a * 2 + q] = yc * WC + xc;
        wq[a * 2 + q] = ok ? wyv[a] * wxv[q] : 0.f;
    }

    const float* p1  = desc1 + (size_t)b * CC * NN + n;
    const float* p2  = desc2 + (size_t)b * CC * NN;
    const float* p2o = p2 + n;
    float s1 = 0.f, s2 = 0.f, ss = 0.f, dr = 0.f;
    for (int c = 0; c < CC; c++) {
        float a = p1[(size_t)c * NN];
        float e = p2o[(size_t)c * NN];
        const float* pc = p2 + (size_t)c * NN;
        float smp = wq[0]*pc[iq[0]] + wq[1]*pc[iq[1]]
                  + wq[2]*pc[iq[2]] + wq[3]*pc[iq[3]];
        s1 += a * a; s2 += e * e; ss += smp * smp; dr += a * smp;
    }
    float rs1 = rsqrtf(s1 + 1e-12f);
    float rs2 = rsqrtf(s2 + 1e-12f);
    float rss = rsqrtf(ss + 1e-12f);
    posA[idx] = 2.f - 2.f * dr * rs1 * rss;

    unsigned* o1 = d1n + (size_t)idx * (CC / 2);
    unsigned* o2 = d2n + (size_t)idx * (CC / 2);
    for (int c = 0; c < CC; c += 2) {
        unsigned l1 = f2bf(p1[(size_t)c * NN] * rs1);
        unsigned h1 = f2bf(p1[(size_t)(c + 1) * NN] * rs1);
        o1[c >> 1] = l1 | (h1 << 16);
        unsigned l2 = f2bf(p2o[(size_t)c * NN] * rs2);
        unsigned h2 = f2bf(p2o[(size_t)(c + 1) * NN] * rs2);
        o2[c >> 1] = l2 | (h2 << 16);
    }
}

// ---------------------------------------------------------------------------
// Kernel 2: per (b, m) visibility: all 64 pixels of frame-2 cell m must warp
// (homo21) to a point where bilinear-sampled ones > 0.  pvis = 5*(1 - vis).
// ---------------------------------------------------------------------------
__global__ __launch_bounds__(256) void vis_kernel(
    const float* __restrict__ homo21, float* __restrict__ pvisA)
{
    int idx = blockIdx.x * blockDim.x + threadIdx.x;
    if (idx >= BB * NN) return;
    int b = idx >> 12, m = idx & (NN - 1);
    const float* H = homo21 + b * 9;
    int cy0 = (m >> 6) * GSZ, cx0 = (m & (WC - 1)) * GSZ;
    bool all = true;
    #pragma unroll 2
    for (int i = 0; i < GSZ; i++)
    for (int j = 0; j < GSZ; j++) {
        float px = (float)(cx0 + j), py = (float)(cy0 + i);
        float wz  = H[6]*px + H[7]*py + H[8] + 1e-8f;
        float inv = 1.f / wz;
        float xw = (H[0]*px + H[1]*py + H[2]) * inv;
        float yw = (H[3]*px + H[4]*py + H[5]) * inv;
        float y0f = floorf(yw), x0f = floorf(xw);
        float fy = yw - y0f, fx = xw - x0f;
        int y0 = (int)y0f, x0 = (int)x0f;
        float v = 0.f;
        v += ((y0 >= 0 && y0 < HH && x0     >= 0 && x0     < WW) ? (1.f-fy)*(1.f-fx) : 0.f);
        v += ((y0 >= 0 && y0 < HH && x0 + 1 >= 0 && x0 + 1 < WW) ? (1.f-fy)*fx       : 0.f);
        v += ((y0+1 >= 0 && y0+1 < HH && x0     >= 0 && x0     < WW) ? fy*(1.f-fx)   : 0.f);
        v += ((y0+1 >= 0 && y0+1 < HH && x0 + 1 >= 0 && x0 + 1 < WW) ? fy*fx         : 0.f);
        all = all && (v > 0.f);
    }
    pvisA[idx] = all ? 0.f : 5.f;
}

// ---------------------------------------------------------------------------
// Kernel 3: fused GEMM (bf16 WMMA) + penalties + row-min, TDM-fed pipeline.
// WG = 128 threads (4 waves); each wave owns TWO 16-row n-tiles (A resident
// in VGPRs -> 2 WMMAs per B-fragment).  D2 tiles (64 rows x 256B) are DMA'd
// into double-buffered LDS by the Tensor Data Mover (one descriptor per tile,
// hardware LDS padding to a 272B stride), overlapped with compute; pvis is
// staged with global_load_async_to_lds_b32 (ASYNCcnt path).
// ---------------------------------------------------------------------------
__global__ __launch_bounds__(128) void simmin_kernel(
    const unsigned* __restrict__ d1n, const unsigned* __restrict__ d2n,
    const float* __restrict__ wyA, const float* __restrict__ wxA,
    const float* __restrict__ pvisA, float* __restrict__ negA)
{
    __shared__ unsigned char ldsD2[2][64 * LDS_STRIDE];
    __shared__ float ldsPvis[2][64];

    int tid  = threadIdx.x;
    int wave = tid >> 5;
    int lane = tid & 31;
    int l16  = lane & 15;
    int hi   = lane >> 4;

    int b    = blockIdx.x >> 5;        // 32 n-blocks (of 128 rows) per batch
    int n0wg = (blockIdx.x & 31) * 128;
    int n0a  = n0wg + wave * 32;       // first 16-row tile
    int n0b  = n0a + 16;               // second 16-row tile

    // A fragments for both tiles: row l16; K split into 4 chunks of 32.
    const char* aRow0 = (const char*)d1n + (size_t)(b * NN + n0a + l16) * 256;
    const char* aRow1 = (const char*)d1n + (size_t)(b * NN + n0b + l16) * 256;
    FragBf aF0[4], aF1[4];
    #pragma unroll
    for (int kk = 0; kk < 4; kk++) {
        aF0[kk].u4[0] = *(const uint4*)(aRow0 + kk * 64 + hi * 16);
        aF0[kk].u4[1] = *(const uint4*)(aRow0 + kk * 64 + 32 + hi * 16);
        aF1[kk].u4[0] = *(const uint4*)(aRow1 + kk * 64 + hi * 16);
        aF1[kk].u4[1] = *(const uint4*)(aRow1 + kk * 64 + 32 + hi * 16);
    }

    // warped coords per accumulator row (loop-invariant over m) -> registers
    float wy0[8], wx0[8], wy1[8], wx1[8];
    #pragma unroll
    for (int v = 0; v < 8; v++) {
        int r0 = b * NN + n0a + v + 8 * hi;
        int r1 = b * NN + n0b + v + 8 * hi;
        wy0[v] = wyA[r0]; wx0[v] = wxA[r0];
        wy1[v] = wyA[r1]; wx1[v] = wxA[r1];
    }

    float rmin0[8], rmin1[8];
    #pragma unroll
    for (int v = 0; v < 8; v++) { rmin0[v] = 3.0e38f; rmin1[v] = 3.0e38f; }

    const char* d2base   = (const char*)d2n + (size_t)b * NN * 256;
    unsigned    ldsD2Lo  = (unsigned)(size_t)&ldsD2[0][0];
    unsigned    ldsPvLo  = (unsigned)(size_t)&ldsPvis[0][0];

    // prologue: stage tile 0 (TDM) + pvis 0 (async-to-LDS)
    if (wave == 0) tdm_load_tile(ldsD2Lo, d2base);
    if (tid < 64) {
        unsigned la = ldsPvLo + (unsigned)(tid * 4);
        const float* ga = pvisA + b * NN + tid;
        asm volatile("global_load_async_to_lds_b32 %0, %1, off"
                     :: "v"(la), "v"(ga) : "memory");
    }
    asm volatile("s_wait_asynccnt 0x0" ::: "memory");

    for (int it = 0; it < NN / 64; ++it) {
        if (wave == 0) __builtin_amdgcn_s_wait_tensorcnt(0);
        __syncthreads();   // tile `it` resident; all waves done with tile it-1

        int nxt = it + 1;
        if (nxt < NN / 64) {
            if (wave == 0)
                tdm_load_tile(ldsD2Lo + (unsigned)((nxt & 1) * (64 * LDS_STRIDE)),
                              d2base + (size_t)nxt * 64 * 256);
            if (tid < 64) {
                unsigned la = ldsPvLo + (unsigned)((nxt & 1) * 256 + tid * 4);
                const float* ga = pvisA + b * NN + nxt * 64 + tid;
                asm volatile("global_load_async_to_lds_b32 %0, %1, off"
                             :: "v"(la), "v"(ga) : "memory");
            }
            __builtin_prefetch(d2base + (size_t)nxt * 64 * 256 + (size_t)tid * 128, 0, 1);
        }

        const unsigned char* buf = ldsD2[it & 1];
        const float*         pvb = ldsPvis[it & 1];
        int m0 = it * 64;

        #pragma unroll
        for (int mt = 0; mt < 4; mt++) {
            const unsigned char* bRow = &buf[(mt * 16 + l16) * LDS_STRIDE];
            v8f acc0 = {};
            v8f acc1 = {};
            #pragma unroll
            for (int kk = 0; kk < 4; kk++) {
                FragBf bF;
                bF.u4[0] = *(const uint4*)(bRow + kk * 64 + hi * 16);
                bF.u4[1] = *(const uint4*)(bRow + kk * 64 + 32 + hi * 16);
                acc0 = __builtin_amdgcn_wmma_f32_16x16x32_bf16(
                    false, aF0[kk].v, false, bF.v, (short)0, acc0, false, false);
                acc1 = __builtin_amdgcn_wmma_f32_16x16x32_bf16(
                    false, aF1[kk].v, false, bF.v, (short)0, acc1, false, false);
            }
            int m = m0 + mt * 16 + l16;
            float cy2 = (float)(m >> 6) * 8.f + 3.5f;
            float cx2 = (float)(m & (WC - 1)) * 8.f + 3.5f;
            float pv  = pvb[mt * 16 + l16];
            #pragma unroll
            for (int v = 0; v < 8; v++) {
                float dy0 = wy0[v] - cy2, dx0 = wx0[v] - cx2;
                float v0 = 2.f - 2.f * acc0[v]
                         + ((dy0 * dy0 + dx0 * dx0 <= 64.f) ? 5.f : 0.f) + pv;
                rmin0[v] = fminf(rmin0[v], v0);
                float dy1 = wy1[v] - cy2, dx1 = wx1[v] - cx2;
                float v1 = 2.f - 2.f * acc1[v]
                         + ((dy1 * dy1 + dx1 * dx1 <= 64.f) ? 5.f : 0.f) + pv;
                rmin1[v] = fminf(rmin1[v], v1);
            }
        }
        // async pvis writes for tile it+1 complete before the next barrier
        asm volatile("s_wait_asynccnt 0x0" ::: "memory");
    }

    // min across the 16 lanes of each half-wave (wave32; xor masks stay in half)
    #pragma unroll
    for (int v = 0; v < 8; v++) {
        #pragma unroll
        for (int mask = 8; mask >= 1; mask >>= 1) {
            rmin0[v] = fminf(rmin0[v], __shfl_xor(rmin0[v], mask, 32));
            rmin1[v] = fminf(rmin1[v], __shfl_xor(rmin1[v], mask, 32));
        }
    }
    if (l16 < 8) {
        float o0 = rmin0[0], o1 = rmin1[0];
        #pragma unroll
        for (int v = 1; v < 8; v++) {
            if (l16 == v) { o0 = rmin0[v]; o1 = rmin1[v]; }
        }
        negA[b * NN + n0a + 8 * hi + l16] = o0;
        negA[b * NN + n0b + 8 * hi + l16] = o1;
    }
}

// ---------------------------------------------------------------------------
// Kernel 4: deterministic single-block reduction -> scalar loss.
// ---------------------------------------------------------------------------
__global__ __launch_bounds__(256) void loss_kernel(
    const float* __restrict__ matchA, const float* __restrict__ posA,
    const float* __restrict__ negA, float* __restrict__ out)
{
    __shared__ float rL[256];
    __shared__ float rM[256];
    int tid = threadIdx.x;
    float sL = 0.f, sM = 0.f;
    for (int i = tid; i < BB * NN; i += 256) {
        float m = matchA[i];
        float l = fmaxf(posA[i] - negA[i] + 1.0f, 0.f);
        sL += m * l * l;
        sM += m;
    }
    rL[tid] = sL; rM[tid] = sM;
    __syncthreads();
    for (int s = 128; s > 0; s >>= 1) {
        if (tid < s) { rL[tid] += rL[tid + s]; rM[tid] += rM[tid + s]; }
        __syncthreads();
    }
    if (tid == 0) out[0] = rL[0] / rM[0];
}

// ---------------------------------------------------------------------------
extern "C" void kernel_launch(void* const* d_in, const int* in_sizes, int n_in,
                              void* d_out, int out_size, void* d_ws, size_t ws_size,
                              hipStream_t stream)
{
    (void)in_sizes; (void)n_in; (void)out_size; (void)ws_size;
    // inputs: score1, score2 (shapes only - never read), desc1, desc2, homo12, homo21
    const float* desc1  = (const float*)d_in[2];
    const float* desc2  = (const float*)d_in[3];
    const float* homo12 = (const float*)d_in[4];
    const float* homo21 = (const float*)d_in[5];

    char* ws = (char*)d_ws;
    size_t off = 0;
    auto take = [&](size_t bytes) {
        char* p = ws + off;
        off += (bytes + 255) & ~(size_t)255;
        return p;
    };
    unsigned* d1n   = (unsigned*)take((size_t)BB * NN * CC * 2);  // bf16 rows
    unsigned* d2n   = (unsigned*)take((size_t)BB * NN * CC * 2);
    float* wyA      = (float*)take((size_t)BB * NN * 4);
    float* wxA      = (float*)take((size_t)BB * NN * 4);
    float* matchA   = (float*)take((size_t)BB * NN * 4);
    float* posA     = (float*)take((size_t)BB * NN * 4);
    float* pvisA    = (float*)take((size_t)BB * NN * 4);
    float* negA     = (float*)take((size_t)BB * NN * 4);

    prep_kernel<<<(BB * NN) / 256, 256, 0, stream>>>(
        desc1, desc2, homo12, d1n, d2n, wyA, wxA, matchA, posA);
    vis_kernel<<<(BB * NN) / 256, 256, 0, stream>>>(homo21, pvisA);
    simmin_kernel<<<BB * 32, 128, 0, stream>>>(
        d1n, d2n, wyA, wxA, pvisA, negA);
    loss_kernel<<<1, 256, 0, stream>>>(matchA, posA, negA, (float*)d_out);
}
